// MultiHeadAttention_87909390614953
// MI455X (gfx1250) — matrix operand order repaired and checked by reading, not compiled, not run on hardware
//
#include <hip/hip_runtime.h>
#include <hip/hip_bf16.h>

// ---------- types ----------
typedef __attribute__((ext_vector_type(16))) __bf16 v16bf;
typedef __attribute__((ext_vector_type(8)))  float  v8f;
typedef unsigned int  u32x4 __attribute__((ext_vector_type(4)));
typedef int           i32x4 __attribute__((ext_vector_type(4)));
typedef int           i32x8 __attribute__((ext_vector_type(8)));

union Frag16 {
  v16bf v;
  unsigned short h[16];
  unsigned int   w[8];
  uint4          q[2];
};

#define NEG_INF (-__builtin_inff())

#define BATCH 2
#define SEQ   2048
#define DM    1024
#define NH    16
#define DH    64
#define MROWS (BATCH * SEQ)   // 4096

#if __has_builtin(__builtin_amdgcn_tensor_load_to_lds)
#define HAVE_TDM 1
#else
#define HAVE_TDM 0
#endif

#if HAVE_TDM
// Issue one TDM 2D tile load: tile is (tile_h rows) x (64 bf16), row stride
// `stride_elems` in global memory. LDS destination gets 8 bf16 of padding per
// row (pad_interval=32 DWORDs, pad_amount=4 DWORDs) -> 72-element row stride,
// matching the ds_load_b128 fragment layout used by the compute code.
__device__ __forceinline__ void tdm_load_2d(unsigned lds_off, const void* gptr,
                                            unsigned tile_h, unsigned stride_elems) {
  unsigned long long ga = (unsigned long long)(size_t)gptr;
  u32x4 g0;
  g0[0] = 1u;                                    // count=1 (valid), user mode
  g0[1] = lds_off;                               // lds_addr (bytes)
  g0[2] = (unsigned)(ga & 0xffffffffu);          // global_addr[31:0]
  g0[3] = (unsigned)((ga >> 32) & 0x01ffffffu)   // global_addr[56:32]
        | 0x80000000u;                           // type = 2 ("image")
  const unsigned td0 = 1u << 30, td1 = 1u << 30; // huge tensor dims: no OOB clip
  const unsigned long long s0 = (unsigned long long)stride_elems;
  i32x8 g1;
  g1[0] = (int)((1u << 16)        // data_size = 2 bytes
              | (1u << 20)        // pad_enable
              | (4u << 22)        // pad_interval: every 32 DWORDs (= one row)
              | (3u << 25));      // pad_amount: 4 DWORDs (= 8 bf16)
  g1[1] = (int)((td0 & 0xffffu) << 16);                  // abar=0 | td0[15:0]
  g1[2] = (int)((td0 >> 16) | ((td1 & 0xffffu) << 16));  // td0[31:16] | td1[15:0]
  g1[3] = (int)((td1 >> 16) | (64u << 16));              // td1[31:16] | tile_dim0=64
  g1[4] = (int)(tile_h & 0xffffu);                       // tile_dim1 | tile_dim2=0
  g1[5] = (int)(unsigned)(s0 & 0xffffffffu);             // stride0[31:0]
  g1[6] = (int)(unsigned)((s0 >> 32) & 0xffffu);         // stride0[47:32] | stride1lo=0
  g1[7] = 0;                                             // stride1 hi (2D: unused)
  i32x4 z4 = {0, 0, 0, 0};
#if defined(__clang_major__) && __clang_major__ >= 23
  i32x8 z8 = {0, 0, 0, 0, 0, 0, 0, 0};
  __builtin_amdgcn_tensor_load_to_lds(g0, g1, z4, z4, z8, 0);
#else
  __builtin_amdgcn_tensor_load_to_lds(g0, g1, z4, z4, 0);
#endif
}
#endif  // HAVE_TDM

// fp32 -> bf16 (round to nearest even), branchless
__device__ __forceinline__ unsigned short f2bf(float f) {
  unsigned int u   = __float_as_uint(f);
  unsigned int rne = u + 0x7fffu + ((u >> 16) & 1u);
  bool         nan = (u & 0x7fffffffu) > 0x7f800000u;
  return (unsigned short)(nan ? ((u >> 16) | 0x40u) : (rne >> 16));
}

// 4-wide fp32 -> bf16 convert (n4 = n/4 vectors)
__global__ void f32_to_bf16_k(const float* __restrict__ in,
                              unsigned short* __restrict__ out, int n4) {
  int i = blockIdx.x * blockDim.x + threadIdx.x;
  if (i < n4) {
    float4 f = ((const float4*)in)[i];
    union { unsigned short h[4]; uint2 u; } o;
    o.h[0] = f2bf(f.x); o.h[1] = f2bf(f.y);
    o.h[2] = f2bf(f.z); o.h[3] = f2bf(f.w);
    ((uint2*)out)[i] = o.u;
  }
}

// ---------------------------------------------------------------------------
// GEMM: C[M,1024] = A[M,1024](bf16) @ B[1024,1024](bf16) + bias(f32), *scale.
// Block: 256 thr (8 waves), tile 128x64, k-step 64. Wave tile 32x32 (2x2 WMMA).
// A tile staged by the Tensor Data Mover (wave 0 issues, TENSORcnt sync);
// B tile staged column-major by all threads so B-frags are aligned b128 loads.
// ---------------------------------------------------------------------------
template <bool WRITE_F32>
__global__ __launch_bounds__(256)
void gemm_bias_bf16(const unsigned short* __restrict__ A,
                    const unsigned short* __restrict__ Bw,
                    const float* __restrict__ bias,
                    float* __restrict__ outF,
                    unsigned short* __restrict__ outB,
                    float scale) {
  __shared__ unsigned short As[128 * 72];   // 128x64 tile, stride 72 (TDM-padded)
  __shared__ unsigned short BsT[64 * 72];   // 64 cols x 64 k (column-major)

  const int tid  = threadIdx.x;
  const int lane = tid & 31;
  const int wave = tid >> 5;
  const int g    = lane >> 4;
  const int ln   = lane & 15;
  const int m0   = blockIdx.y * 128;
  const int n0   = blockIdx.x * 64;
  const int wm   = (wave >> 1) * 32;
  const int wn   = (wave & 1) * 32;

  v8f acc[2][2];
  acc[0][0] = {}; acc[0][1] = {}; acc[1][0] = {}; acc[1][1] = {};

  for (int k0 = 0; k0 < DM; k0 += 64) {
    __syncthreads();
#if HAVE_TDM
    if (wave == 0)  // wave-uniform: one TDM DMA fills the whole A tile
      tdm_load_2d((unsigned)(size_t)&As[0], A + (size_t)m0 * DM + k0, 128, DM);
#else
#pragma unroll
    for (int s = 0; s < 4; ++s) {
      int seg = tid + s * 256;
      int r = seg >> 3, c = (seg & 7) << 3;
      *(uint4*)&As[r * 72 + c] = *(const uint4*)&A[(size_t)(m0 + r) * DM + k0 + c];
    }
#endif
    // stage B tile transposed: 64k x 64n -> BsT[col][k]
#pragma unroll
    for (int s = 0; s < 2; ++s) {
      int seg = tid + s * 256;
      int r = seg >> 3, c = (seg & 7) << 3;  // r = k row, c = col base
      uint4 d = *(const uint4*)&Bw[(size_t)(k0 + r) * DM + n0 + c];
      const unsigned short* de = (const unsigned short*)&d;
#pragma unroll
      for (int e = 0; e < 8; ++e) BsT[(c + e) * 72 + r] = de[e];
    }
#if HAVE_TDM
    if (wave == 0) __builtin_amdgcn_s_wait_tensorcnt(0);
#endif
    __syncthreads();

#pragma unroll
    for (int kc = 0; kc < 2; ++kc) {
      Frag16 a[2], b[2];
#pragma unroll
      for (int i = 0; i < 2; ++i) {
        const unsigned short* ap = &As[(wm + i * 16 + ln) * 72 + kc * 32 + (g << 3)];
        a[i].q[0] = *(const uint4*)ap;        // K = 32kc + 8g + 0..7
        a[i].q[1] = *(const uint4*)(ap + 16); // K = 32kc + 16 + 8g + 0..7
      }
#pragma unroll
      for (int j = 0; j < 2; ++j) {
        const unsigned short* bp = &BsT[(wn + j * 16 + ln) * 72 + kc * 32 + (g << 3)];
        b[j].q[0] = *(const uint4*)bp;        // K = 32kc + 8g + 0..7
        b[j].q[1] = *(const uint4*)(bp + 16); // K = 32kc + 16 + 8g + 0..7
      }
#pragma unroll
      for (int i = 0; i < 2; ++i)
#pragma unroll
        for (int j = 0; j < 2; ++j)
          acc[i][j] = __builtin_amdgcn_wmma_f32_16x16x32_bf16(
              false, a[i].v, false, b[j].v, (short)0, acc[i][j], false, false);
    }
  }

  // epilogue: C/D element (M = r + 8g, N = lane%16)
#pragma unroll
  for (int j = 0; j < 2; ++j) {
    const int   col = n0 + wn + j * 16 + ln;
    const float bv_ = bias[col];
#pragma unroll
    for (int i = 0; i < 2; ++i) {
      size_t base = (size_t)(m0 + wm + i * 16 + (g << 3)) * DM + col;
#pragma unroll
      for (int r = 0; r < 8; ++r) {
        float val = (acc[i][j][r] + bv_) * scale;
        if constexpr (WRITE_F32) outF[base + (size_t)r * DM] = val;
        else                     outB[base + (size_t)r * DM] = f2bf(val);
      }
    }
  }
}

// ---------------------------------------------------------------------------
// Flash attention. Block: 128 thr (4 waves), 64 query rows, key tiles of 64.
// grid.x = SEQ/64, grid.y = BATCH*NH. Q pre-scaled by 1/sqrt(Dh).
// K tile staged by TDM; V tile staged transposed by all threads.
// ---------------------------------------------------------------------------
__global__ __launch_bounds__(128)
void flash_attn(const unsigned short* __restrict__ Q,
                const unsigned short* __restrict__ Kg,
                const unsigned short* __restrict__ Vg,
                const unsigned char* __restrict__ mask,
                unsigned short* __restrict__ Ctx) {
  __shared__ unsigned short Ks[64 * 72];   // [key][d]   (TDM-padded stride 72)
  __shared__ unsigned short VsT[64 * 72];  // [d][key]   (transposed)
  __shared__ unsigned short Ps[64 * 72];   // probabilities, [query][key]

  const int tid  = threadIdx.x;
  const int lane = tid & 31;
  const int wave = tid >> 5;
  const int g    = lane >> 4;
  const int ln   = lane & 15;
  const int b    = blockIdx.y >> 4;   // NH = 16
  const int h    = blockIdx.y & 15;
  const int q0   = blockIdx.x * 64;
  const int hoff = h * DH;

  // resident Q A-fragments (16 rows x 64 d), two k-chunks of 32
  Frag16 aq[2];
  {
    const unsigned short* qrow =
        &Q[(size_t)(b * SEQ + q0 + wave * 16 + ln) * DM + hoff];
#pragma unroll
    for (int kc = 0; kc < 2; ++kc) {
      aq[kc].q[0] = *(const uint4*)(qrow + kc * 32 + (g << 3));
      aq[kc].q[1] = *(const uint4*)(qrow + kc * 32 + 16 + (g << 3));
    }
  }

  float m_i[8], l_i[8];
  v8f acc[4];
#pragma unroll
  for (int r = 0; r < 8; ++r) { m_i[r] = NEG_INF; l_i[r] = 0.f; }
  acc[0] = {}; acc[1] = {}; acc[2] = {}; acc[3] = {};

  for (int j0 = 0; j0 < SEQ; j0 += 64) {
    __syncthreads();  // protect K/V LDS reuse across iterations
#if HAVE_TDM
    if (wave == 0)  // DMA the K tile (64x64 bf16) into LDS, padded to stride 72
      tdm_load_2d((unsigned)(size_t)&Ks[0],
                  Kg + (size_t)(b * SEQ + j0) * DM + hoff, 64, DM);
#endif
    // stage V tile transposed (and K manually if no TDM)
#pragma unroll
    for (int s = 0; s < 4; ++s) {
      int seg = tid + s * 128;
      int r = seg >> 3, c = (seg & 7) << 3;
      size_t gidx = (size_t)(b * SEQ + j0 + r) * DM + hoff + c;
#if !HAVE_TDM
      *(uint4*)&Ks[r * 72 + c] = *(const uint4*)&Kg[gidx];
#endif
      uint4 vv = *(const uint4*)&Vg[gidx];
      const unsigned short* ve = (const unsigned short*)&vv;
#pragma unroll
      for (int e = 0; e < 8; ++e) VsT[(c + e) * 72 + r] = ve[e];
    }
    if (j0 + 64 < SEQ) {  // hint next tiles into cache (global_prefetch_b8)
      size_t nidx = (size_t)(b * SEQ + j0 + 64 + (tid >> 1)) * DM + hoff + (tid & 1) * 32;
      __builtin_prefetch(&Kg[nidx], 0, 1);
      __builtin_prefetch(&Vg[nidx], 0, 1);
    }
#if HAVE_TDM
    if (wave == 0) __builtin_amdgcn_s_wait_tensorcnt(0);
#endif
    __syncthreads();

    // S = Q @ K^T : contraction over d (2 chunks), N = key (4 subtiles)
    v8f sc[4];
    sc[0] = {}; sc[1] = {}; sc[2] = {}; sc[3] = {};
#pragma unroll
    for (int nt = 0; nt < 4; ++nt) {
      const unsigned short* krow = &Ks[(nt * 16 + ln) * 72];
#pragma unroll
      for (int kc = 0; kc < 2; ++kc) {
        Frag16 bk;
        bk.q[0] = *(const uint4*)(krow + kc * 32 + (g << 3));
        bk.q[1] = *(const uint4*)(krow + kc * 32 + 16 + (g << 3));
        sc[nt] = __builtin_amdgcn_wmma_f32_16x16x32_bf16(
            false, aq[kc].v, false, bk.v, (short)0, sc[nt], false, false);
      }
    }

    // key-position mask (True = masked out)
    unsigned char mk[4];
#pragma unroll
    for (int nt = 0; nt < 4; ++nt) mk[nt] = mask[b * SEQ + j0 + nt * 16 + ln];

    // online softmax over rows M = r + 8g (16 lanes of a half hold one row)
    float alpha[8];
#pragma unroll
    for (int r = 0; r < 8; ++r) {
#pragma unroll
      for (int nt = 0; nt < 4; ++nt)
        if (mk[nt]) sc[nt][r] = NEG_INF;
      float mx = NEG_INF;
#pragma unroll
      for (int nt = 0; nt < 4; ++nt) mx = fmaxf(mx, sc[nt][r]);
      mx = fmaxf(mx, __shfl_xor(mx, 1, 32));
      mx = fmaxf(mx, __shfl_xor(mx, 2, 32));
      mx = fmaxf(mx, __shfl_xor(mx, 4, 32));
      mx = fmaxf(mx, __shfl_xor(mx, 8, 32));
      float mn = fmaxf(m_i[r], mx);
      float al = (mn == NEG_INF) ? 1.f : __expf(m_i[r] - mn);
      float rs = 0.f;
#pragma unroll
      for (int nt = 0; nt < 4; ++nt) {
        float p = (sc[nt][r] == NEG_INF) ? 0.f : __expf(sc[nt][r] - mn);
        sc[nt][r] = p;
        rs += p;
      }
      rs += __shfl_xor(rs, 1, 32);
      rs += __shfl_xor(rs, 2, 32);
      rs += __shfl_xor(rs, 4, 32);
      rs += __shfl_xor(rs, 8, 32);
      l_i[r] = l_i[r] * al + rs;
      m_i[r] = mn;
      alpha[r] = al;
    }
#pragma unroll
    for (int dt = 0; dt < 4; ++dt)
#pragma unroll
      for (int r = 0; r < 8; ++r) acc[dt][r] *= alpha[r];

    // spill P (C/D layout) to LDS so it can be reloaded in A-frag layout
#pragma unroll
    for (int nt = 0; nt < 4; ++nt)
#pragma unroll
      for (int r = 0; r < 8; ++r)
        Ps[(wave * 16 + r + (g << 3)) * 72 + nt * 16 + ln] = f2bf(sc[nt][r]);
    asm volatile("s_wait_dscnt 0" ::: "memory");  // wave-local LDS WAR/RAW fence

    // ctx += P @ V : contraction over key (2 chunks), N = d (4 subtiles)
#pragma unroll
    for (int kc = 0; kc < 2; ++kc) {
      Frag16 ap;
      const unsigned short* prow = &Ps[(wave * 16 + ln) * 72 + kc * 32];
      ap.q[0] = *(const uint4*)(prow + (g << 3));
      ap.q[1] = *(const uint4*)(prow + 16 + (g << 3));
#pragma unroll
      for (int dt = 0; dt < 4; ++dt) {
        Frag16 bv;
        const unsigned short* vrow = &VsT[(dt * 16 + ln) * 72 + kc * 32];
        bv.q[0] = *(const uint4*)(vrow + (g << 3));
        bv.q[1] = *(const uint4*)(vrow + 16 + (g << 3));
        acc[dt] = __builtin_amdgcn_wmma_f32_16x16x32_bf16(
            false, ap.v, false, bv.v, (short)0, acc[dt], false, false);
      }
    }
  }

  // normalize and write ctx (merged-head layout, bf16 for the output GEMM)
#pragma unroll
  for (int r = 0; r < 8; ++r) {
    float l = l_i[r];
    float inv = (l > 0.f) ? 1.f / l : 0.f;
    size_t row = (size_t)(b * SEQ + q0 + wave * 16 + r + (g << 3));
#pragma unroll
    for (int dt = 0; dt < 4; ++dt)
      Ctx[row * DM + hoff + dt * 16 + ln] = f2bf(acc[dt][r] * inv);
  }
}

// ---------------------------------------------------------------------------
extern "C" void kernel_launch(void* const* d_in, const int* in_sizes, int n_in,
                              void* d_out, int out_size, void* d_ws, size_t ws_size,
                              hipStream_t stream) {
  (void)in_sizes; (void)n_in; (void)out_size; (void)ws_size;
  const float* x  = (const float*)d_in[0];
  const float* Wq = (const float*)d_in[1];
  const float* bq = (const float*)d_in[2];
  const float* Wk = (const float*)d_in[3];
  const float* bk = (const float*)d_in[4];
  const float* Wv = (const float*)d_in[5];
  const float* bv = (const float*)d_in[6];
  const float* Wo = (const float*)d_in[7];
  const float* bo = (const float*)d_in[8];
  const unsigned char* mask = (const unsigned char*)d_in[9];  // jnp bool -> 1 byte

  char* ws = (char*)d_ws;
  size_t off = 0;
  auto carve = [&](size_t elems) {
    void* p = ws + off;
    off = (off + elems * sizeof(unsigned short) + 255) & ~(size_t)255;
    return (unsigned short*)p;
  };
  unsigned short* xb  = carve((size_t)MROWS * DM);
  unsigned short* wqb = carve((size_t)DM * DM);
  unsigned short* wkb = carve((size_t)DM * DM);
  unsigned short* wvb = carve((size_t)DM * DM);
  unsigned short* wob = carve((size_t)DM * DM);
  unsigned short* qb  = carve((size_t)MROWS * DM);
  unsigned short* kb  = carve((size_t)MROWS * DM);
  unsigned short* vb  = carve((size_t)MROWS * DM);
  unsigned short* ctx = carve((size_t)MROWS * DM);

  const int nx4 = MROWS * DM / 4, nw4 = DM * DM / 4;
  f32_to_bf16_k<<<(nx4 + 255) / 256, 256, 0, stream>>>(x,  xb,  nx4);
  f32_to_bf16_k<<<(nw4 + 255) / 256, 256, 0, stream>>>(Wq, wqb, nw4);
  f32_to_bf16_k<<<(nw4 + 255) / 256, 256, 0, stream>>>(Wk, wkb, nw4);
  f32_to_bf16_k<<<(nw4 + 255) / 256, 256, 0, stream>>>(Wv, wvb, nw4);
  f32_to_bf16_k<<<(nw4 + 255) / 256, 256, 0, stream>>>(Wo, wob, nw4);

  dim3 gg(DM / 64, MROWS / 128);
  // Q projection with softmax scale folded in (1/sqrt(64) = 0.125)
  gemm_bias_bf16<false><<<gg, 256, 0, stream>>>(xb, wqb, bq, nullptr, qb, 0.125f);
  gemm_bias_bf16<false><<<gg, 256, 0, stream>>>(xb, wkb, bk, nullptr, kb, 1.0f);
  gemm_bias_bf16<false><<<gg, 256, 0, stream>>>(xb, wvb, bv, nullptr, vb, 1.0f);

  dim3 ga(SEQ / 64, BATCH * NH);
  flash_attn<<<ga, 128, 0, stream>>>(qb, kb, vb, mask, ctx);

  gemm_bias_bf16<true><<<gg, 256, 0, stream>>>(ctx, wob, bo, (float*)d_out, nullptr, 1.0f);
}
